// SinkhornOTLoss_41386304864328
// MI455X (gfx1250) — matrix operand order repaired and checked by reading, not compiled, not run on hardware
//
#include <hip/hip_runtime.h>
#include <math.h>

// Sinkhorn OT loss for MI455X (gfx1250, wave32, WMMA).
// pred/target: [64, 256, 64, 64] f32.  Output: 1 f32 (mean loss).
// ws usage: dot matrices 64*256*256 f32 (16 MB) + 64 f32 losses.
//
// Phase 2 runs Sinkhorn in the kernel (exp) domain: K = exp(-cost/eps) computed
// once, then a = mu/(K b), b = nu/(K^T a) — algebraically identical to the
// reference's log-domain iteration (a=e^u, b=e^v), but 60 matvecs of pure FMA
// instead of 60 logsumexp passes (saves ~4M v_exp_f32 per workgroup and halves
// LDS traffic). f32 range is safe: K in [e^-40, 1], a,b ~ O(10) here.

#define NB    64      // batches
#define CH    256     // channels
#define HWSZ  4096    // H*W
#define TK    256     // tokens (== MAX_TOKENS)
#define LDH   264     // GEMM LDS row in halves: 528B, 16B aligned, bank-rotating
#define KLD   260     // Sinkhorn K row in floats: 16B-aligned rows, 260%64=4 rotation
#define EPS   0.05f
#define MUV   (1.0f/256.0f)
#define ITERS 30

typedef __attribute__((ext_vector_type(16))) _Float16 v16h;
typedef __attribute__((ext_vector_type(8)))  _Float16 v8h;
typedef __attribute__((ext_vector_type(8)))  float    v8f;

// A fragment (16x32 f16): lane L holds row M=L%16; halves 0..7 -> K = kc + 8*(L/16)+j,
// halves 8..15 -> K = kc + 16 + 8*(L/16) + (j-8).   (ISA 7.12.2, 16-bit A 16x32)
__device__ __forceinline__ v16h frag_a(const _Float16* __restrict__ row, int kc, int lane) {
  const int kh = (lane >> 4) * 8;
  v8h lo = *(const v8h*)(row + kc + kh);
  v8h hi = *(const v8h*)(row + kc + 16 + kh);
  return __builtin_shufflevector(lo, hi, 0,1,2,3,4,5,6,7,8,9,10,11,12,13,14,15);
}

// B fragment (32x16 f16): lane L holds col N=L%16; halves 0..7 -> K = kc + 8*(L/16)+j,
// halves 8..15 -> K = kc + 16 + 8*(L/16) + (j-8).
// B[k][n] = t[n][k] (t stored row-major by token), giving p @ t^T.
__device__ __forceinline__ v16h frag_b(const _Float16* __restrict__ row, int kc, int lane) {
  const int kb = (lane >> 4) * 8;
  v8h lo = *(const v8h*)(row + kc + kb);
  v8h hi = *(const v8h*)(row + kc + 16 + kb);
  return __builtin_shufflevector(lo, hi, 0,1,2,3,4,5,6,7,8,9,10,11,12,13,14,15);
}

// ---------------- Kernel 1: gather + normalize + WMMA GEMM -> dot matrix ----------------
__global__ void __launch_bounds__(256)
tokens_gemm_kernel(const float* __restrict__ pred, const float* __restrict__ targ,
                   float* __restrict__ dot_out) {
  __shared__ _Float16 p_lds[TK][LDH];   // 135168 B
  __shared__ _Float16 t_lds[TK][LDH];   // 135168 B
  __shared__ float invp[TK], invt[TK];  // 2 KB   (total ~266 KB < 320 KB WGP LDS)

  const int b   = blockIdx.x;
  const int tid = threadIdx.x;               // token index 0..255
  const size_t base = (size_t)b * CH * HWSZ;
  const float* __restrict__ pb = pred + base;
  const float* __restrict__ tb = targ + base;

  // linspace(0, 4095, 256) truncated to int
  const int hw = (tid * (HWSZ - 1)) / (TK - 1);

  float ssp = 0.f, sst = 0.f;
  #pragma unroll 4
  for (int c = 0; c < CH; ++c) {
    // strided gather: one 16KB plane per channel; prefetch ahead (global_prefetch_b8)
    __builtin_prefetch(pb + (size_t)(c + 16) * HWSZ + hw, 0, 1);
    __builtin_prefetch(tb + (size_t)(c + 16) * HWSZ + hw, 0, 1);
    float vp = pb[(size_t)c * HWSZ + hw];
    float vt = tb[(size_t)c * HWSZ + hw];
    ssp += vp * vp;
    sst += vt * vt;
    p_lds[tid][c] = (_Float16)vp;           // store raw; normalization folded post-GEMM
    t_lds[tid][c] = (_Float16)vt;
  }
  invp[tid] = 1.0f / fmaxf(sqrtf(ssp), 1e-12f);
  invt[tid] = 1.0f / fmaxf(sqrtf(sst), 1e-12f);
  __syncthreads();

  const int wave = tid >> 5;                 // 8 waves
  const int lane = tid & 31;
  float* __restrict__ dot_b = dot_out + (size_t)b * TK * TK;

  // Each wave: 2 M-tiles x 16 N-tiles of 16x16; A-fragments reused across N loop.
  for (int mi = 0; mi < 2; ++mi) {
    const int mt = wave * 2 + mi;
    const _Float16* arow = &p_lds[mt * 16 + (lane & 15)][0];
    v16h af[8];
    #pragma unroll
    for (int kc = 0; kc < 8; ++kc) af[kc] = frag_a(arow, kc * 32, lane);

    for (int nt = 0; nt < 16; ++nt) {
      const _Float16* brow = &t_lds[nt * 16 + (lane & 15)][0];
      v8f acc = {};
      #pragma unroll
      for (int kc = 0; kc < 8; ++kc) {
        v16h bf = frag_b(brow, kc * 32, lane);
        acc = __builtin_amdgcn_wmma_f32_16x16x32_f16(
            /*neg_a=*/false, af[kc], /*neg_b=*/false, bf,
            /*c_mod=*/(short)0, acc, /*reuse_a=*/false, /*reuse_b=*/false);
      }
      // C layout: VGPR r -> M = r + 8*(lane/16); N = lane%16
      const int n  = nt * 16 + (lane & 15);
      const float sn = invt[n];
      #pragma unroll
      for (int r = 0; r < 8; ++r) {
        const int m = mt * 16 + r + ((lane >> 4) << 3);
        dot_b[(size_t)m * TK + n] = acc[r] * invp[m] * sn;  // normalized cosine dot
      }
    }
  }
}

// ---------------- Kernel 2: exp-domain Sinkhorn, K LDS-resident, 32 waves ----------------
__global__ void __launch_bounds__(1024)
sinkhorn_kernel(const float* __restrict__ dot_in, float* __restrict__ loss_out) {
  __shared__ float K[TK][KLD];          // 266240 B; rows 16B-aligned, banks rotate by 4/row
  __shared__ float psum[4][KLD];        // 4160 B partial combine
  __shared__ float a_s[TK], b_s[TK];    // 2 KB
  __shared__ float part[1024];          // 4 KB     (total ~277 KB < 320 KB)

  const int b   = blockIdx.x;
  const int tid = threadIdx.x;
  const int r   = tid & 255;            // row / column index
  const int q   = tid >> 8;             // 0..3 segment of the reduction axis
  const float* __restrict__ db = dot_in + (size_t)b * TK * TK;

  // K = exp(log_kernel) = exp((dot-1)/eps); the only exp pass (65536 exps, once).
  for (int i = q; i < TK; i += 4)
    K[i][r] = __expf((1.0f / EPS) * (db[(size_t)i * TK + r] - 1.0f));
  if (q == 0) b_s[r] = 1.0f;            // b = exp(v0) = 1
  __syncthreads();

  for (int it = 0; it < ITERS; ++it) {
    // a = mu / (K b)   (row scans: float4 ds_load_b128, 64 cols per thread)
    {
      const float* Krow = &K[r][q * 64];
      const float* bseg = &b_s[q * 64];
      float p = 0.f;
      #pragma unroll
      for (int j = 0; j < 64; j += 4) {
        float4 kv = *(const float4*)(Krow + j);
        float4 bv = *(const float4*)(bseg + j);
        p += kv.x * bv.x + kv.y * bv.y + kv.z * bv.z + kv.w * bv.w;
      }
      psum[q][r] = p;
    }
    __syncthreads();
    if (q == 0) {
      float y = psum[0][r] + psum[1][r] + psum[2][r] + psum[3][r];
      a_s[r] = MUV / fmaxf(y, 1e-35f);
    }
    __syncthreads();
    // b = nu / (K^T a)   (column scans: stride KLD -> conflict-free)
    {
      float p = 0.f;
      #pragma unroll 4
      for (int m = q * 64; m < q * 64 + 64; ++m)
        p += K[m][r] * a_s[m];
      psum[q][r] = p;
    }
    __syncthreads();
    if (q == 0) {
      float y = psum[0][r] + psum[1][r] + psum[2][r] + psum[3][r];
      b_s[r] = MUV / fmaxf(y, 1e-35f);
    }
    __syncthreads();
  }

  // loss = sum_{m,n} a[m]*K[m][n]*b[n] * cost[m][n], cost = 1 - dot (re-read, L2-hot,
  // coalesced: lanes sweep column index r).
  {
    const float bc = b_s[r];
    float p = 0.f;
    #pragma unroll 4
    for (int m = q * 64; m < q * 64 + 64; ++m) {
      const float t = a_s[m] * K[m][r] * bc;           // transport
      p += t * (1.0f - db[(size_t)m * TK + r]);        // * cost
    }
    part[tid] = p;
  }
  __syncthreads();
  for (int off = 512; off > 0; off >>= 1) {
    if (tid < off) part[tid] += part[tid + off];
    __syncthreads();
  }
  if (tid == 0) loss_out[b] = part[0];
}

// ---------------- Kernel 3: deterministic mean over batches ----------------
__global__ void __launch_bounds__(64)
reduce_kernel(const float* __restrict__ loss_in, float* __restrict__ out) {
  __shared__ float r[NB];
  const int t = threadIdx.x;
  r[t] = loss_in[t];
  __syncthreads();
  for (int off = 32; off > 0; off >>= 1) {
    if (t < off) r[t] += r[t + off];
    __syncthreads();
  }
  if (t == 0) out[0] = r[0] * (1.0f / (float)NB);
}

extern "C" void kernel_launch(void* const* d_in, const int* in_sizes, int n_in,
                              void* d_out, int out_size, void* d_ws, size_t ws_size,
                              hipStream_t stream) {
  const float* pred = (const float*)d_in[0];
  const float* targ = (const float*)d_in[1];
  float* out = (float*)d_out;

  float* dot  = (float*)d_ws;                         // 64*256*256 f32 = 16 MB
  float* loss = dot + (size_t)NB * TK * TK;           // 64 f32

  tokens_gemm_kernel<<<NB, 256, 0, stream>>>(pred, targ, dot);
  sinkhorn_kernel<<<NB, 1024, 0, stream>>>(dot, loss);
  reduce_kernel<<<1, 64, 0, stream>>>(loss, out);
}
